// PolicyNet2_89352499626162
// MI455X (gfx1250) — compile-verified, matrix operand order checked by reading, not run on hardware
//
#include <hip/hip_runtime.h>

// ---------------------------------------------------------------------------
// CDNA5 (gfx1250) implementation of PolicyNet2 forward.
// All large GEMMs run on v_wmma_f32_16x16x32_f16 (fp32 data converted to f16
// while staging tiles into LDS; fp32 accumulation). wave32, 8 waves / block,
// double-buffered LDS stages so global-load latency hides under the WMMA chain.
// ---------------------------------------------------------------------------

typedef __attribute__((ext_vector_type(16))) _Float16 v16h;
typedef __attribute__((ext_vector_type(8)))  _Float16 v8h;
typedef __attribute__((ext_vector_type(8)))  float    v8f;

#define WMMA_F16(a, b, c) \
  __builtin_amdgcn_wmma_f32_16x16x32_f16(false, (a), false, (b), (short)0, (c), false, false)

// Load a 16x32 (rows x K) f16 fragment from LDS following the ISA 7.12.2
// 16-bit A/B layout: lanes 0-15 -> row = lane, K halves {0..7,16..23};
// lanes 16-31 -> row = lane-16, K halves {8..15,24..31}.
// Rows are K-contiguous in LDS, so this is two ds_load_b128 per lane.
__device__ __forceinline__ v16h lds_frag(const _Float16* base, int ld) {
  const int lane = threadIdx.x & 31;
  const int r  = lane & 15;
  const int kh = (lane & 16) ? 8 : 0;
  const _Float16* p = base + r * ld + kh;
  v8h lo = *(const v8h*)(p);
  v8h hi = *(const v8h*)(p + 16);
  v16h o;
#pragma unroll
  for (int i = 0; i < 8; ++i) { o[i] = lo[i]; o[i + 8] = hi[i]; }
  return o;
}

// ---------------------------------------------------------------------------
// Generic tiled WMMA GEMM:  C[M,N] = act(A[M,K] * W[N,K]^T + bias) (+resid)
// BN=128, BK=32. BM=128 -> 4x2 wave grid, wave tile 32x64 (8 WMMA/step);
// BM=64 -> 2x4 wave grid, wave tile 32x32 (4 WMMA/step).
// Double-buffered LDS: global loads for tile k+1 are issued before the WMMAs
// of tile k; one barrier per K-step.
// flags: bit0 = ReLU, bit1 = split-K atomic accumulate (raw, no bias/act).
// VEC=4 uses float4 staging (requires lda%4==0 && ldw%4==0), VEC=1 scalar.
// ---------------------------------------------------------------------------
template <int VEC, int BM>
__global__ __launch_bounds__(256) void wmma_gemm(
    const float* __restrict__ A, long lda,
    const float* __restrict__ W, long ldw,
    const float* __restrict__ bias,
    const float* __restrict__ resid,
    float* __restrict__ C, long ldc,
    int M, int N, int K, int k0, int kspan, int flags)
{
  constexpr int BN = 128, BK = 32, LD = 40;     // LD=40 halves: bank-pad
  constexpr int WAVES_M = BM / 32;              // 4 (BM=128) or 2 (BM=64)
  constexpr int WAVES_N = 8 / WAVES_M;          // 2 or 4
  constexpr int TN      = BN / (WAVES_N * 16);  // 4 or 2 n-tiles per wave
  constexpr int A_F4 = BM * 8 / 256;            // float4/thread for A stage
  constexpr int W_F4 = BN * 8 / 256;            // float4/thread for W stage
  constexpr int A_SC = BM * 32 / 256;           // scalars/thread for A stage
  constexpr int W_SC = BN * 32 / 256;

  __shared__ _Float16 sA[2][BM * LD];
  __shared__ _Float16 sW[2][BN * LD];

  const int tid  = threadIdx.x;
  const int w    = tid >> 5;
  const int lane = tid & 31;
  const int wm = w / WAVES_N;
  const int wn = w % WAVES_N;
  const int bm0 = blockIdx.y * BM;
  const int bn0 = blockIdx.x * BN;
  int kbeg = k0 + blockIdx.z * kspan;
  int kend = kbeg + kspan; if (kend > K) kend = K;

  const v8f vz = {0.f,0.f,0.f,0.f,0.f,0.f,0.f,0.f};
  v8f acc[2][TN];
#pragma unroll
  for (int tm = 0; tm < 2; ++tm)
#pragma unroll
    for (int tn = 0; tn < TN; ++tn) acc[tm][tn] = vz;

  float areg[VEC == 4 ? A_F4 * 4 : A_SC];
  float wreg[VEC == 4 ? W_F4 * 4 : W_SC];

  auto load_tile = [&](int kk) {
    if (VEC == 4) {
#pragma unroll
      for (int i = 0; i < A_F4; ++i) {
        int f = tid + i * 256;
        int row = f >> 3, c = (f & 7) << 2;
        int gm = bm0 + row, gk = kk + c;
        float4 val = make_float4(0.f, 0.f, 0.f, 0.f);
        if (gm < M) {
          if (gk + 3 < kend) {
            val = *(const float4*)(A + (long)gm * lda + gk);
          } else {
            float t0 = (gk+0 < kend) ? A[(long)gm*lda+gk+0] : 0.f;
            float t1 = (gk+1 < kend) ? A[(long)gm*lda+gk+1] : 0.f;
            float t2 = (gk+2 < kend) ? A[(long)gm*lda+gk+2] : 0.f;
            float t3 = (gk+3 < kend) ? A[(long)gm*lda+gk+3] : 0.f;
            val = make_float4(t0, t1, t2, t3);
          }
        }
        areg[i*4+0] = val.x; areg[i*4+1] = val.y;
        areg[i*4+2] = val.z; areg[i*4+3] = val.w;
      }
#pragma unroll
      for (int i = 0; i < W_F4; ++i) {
        int f = tid + i * 256;
        int row = f >> 3, c = (f & 7) << 2;
        int gw = bn0 + row, gk = kk + c;
        float4 val = make_float4(0.f, 0.f, 0.f, 0.f);
        if (gw < N) {
          if (gk + 3 < kend) {
            val = *(const float4*)(W + (long)gw * ldw + gk);
          } else {
            float t0 = (gk+0 < kend) ? W[(long)gw*ldw+gk+0] : 0.f;
            float t1 = (gk+1 < kend) ? W[(long)gw*ldw+gk+1] : 0.f;
            float t2 = (gk+2 < kend) ? W[(long)gw*ldw+gk+2] : 0.f;
            float t3 = (gk+3 < kend) ? W[(long)gw*ldw+gk+3] : 0.f;
            val = make_float4(t0, t1, t2, t3);
          }
        }
        wreg[i*4+0] = val.x; wreg[i*4+1] = val.y;
        wreg[i*4+2] = val.z; wreg[i*4+3] = val.w;
      }
    } else {
#pragma unroll
      for (int i = 0; i < A_SC; ++i) {
        int e = tid + i * 256;
        int row = e >> 5, c = e & 31;
        int gm = bm0 + row, gk = kk + c;
        areg[i] = (gm < M && gk < kend) ? A[(long)gm*lda + gk] : 0.f;
      }
#pragma unroll
      for (int i = 0; i < W_SC; ++i) {
        int e = tid + i * 256;
        int row = e >> 5, c = e & 31;
        int gw = bn0 + row, gk = kk + c;
        wreg[i] = (gw < N && gk < kend) ? W[(long)gw*ldw + gk] : 0.f;
      }
    }
  };

  auto store_tile = [&](int buf) {
    if (VEC == 4) {
#pragma unroll
      for (int i = 0; i < A_F4; ++i) {
        int f = tid + i * 256;
        int row = f >> 3, c = (f & 7) << 2;
        _Float16* d = &sA[buf][row * LD + c];
        d[0] = (_Float16)areg[i*4+0]; d[1] = (_Float16)areg[i*4+1];
        d[2] = (_Float16)areg[i*4+2]; d[3] = (_Float16)areg[i*4+3];
      }
#pragma unroll
      for (int i = 0; i < W_F4; ++i) {
        int f = tid + i * 256;
        int row = f >> 3, c = (f & 7) << 2;
        _Float16* d = &sW[buf][row * LD + c];
        d[0] = (_Float16)wreg[i*4+0]; d[1] = (_Float16)wreg[i*4+1];
        d[2] = (_Float16)wreg[i*4+2]; d[3] = (_Float16)wreg[i*4+3];
      }
    } else {
#pragma unroll
      for (int i = 0; i < A_SC; ++i) {
        int e = tid + i * 256;
        sA[buf][(e >> 5) * LD + (e & 31)] = (_Float16)areg[i];
      }
#pragma unroll
      for (int i = 0; i < W_SC; ++i) {
        int e = tid + i * 256;
        sW[buf][(e >> 5) * LD + (e & 31)] = (_Float16)wreg[i];
      }
    }
  };

  // prologue: stage first tile
  load_tile(kbeg);
  store_tile(0);
  __syncthreads();

  int cur = 0;
  for (int kk = kbeg; kk < kend; kk += BK) {
    const bool has_next = (kk + BK) < kend;
    if (has_next) load_tile(kk + BK);          // global loads in flight...
    // prefetch two tiles ahead on the dominant (weight) stream
    {
      int pr = bn0 + (tid >> 1);
      if (pr < N && kk + 2 * BK < kend)
        __builtin_prefetch(W + (long)pr * ldw + kk + 2 * BK, 0, 1);
    }
    // ...while the WMMA chain runs on the current LDS stage
    v16h a0 = lds_frag(&sA[cur][(wm*32 +  0) * LD], LD);
    v16h a1 = lds_frag(&sA[cur][(wm*32 + 16) * LD], LD);
#pragma unroll
    for (int tn = 0; tn < TN; ++tn) {
      v16h bf = lds_frag(&sW[cur][(wn*(TN*16) + tn*16) * LD], LD);
      acc[0][tn] = WMMA_F16(a0, bf, acc[0][tn]);
      acc[1][tn] = WMMA_F16(a1, bf, acc[1][tn]);
    }
    if (has_next) store_tile(cur ^ 1);         // write the *other* stage
    __syncthreads();
    cur ^= 1;
  }

  // Epilogue. C/D layout: element r -> M = r + (lane&16?8:0), N = lane&15.
  const int hh  = (lane & 16) ? 8 : 0;
  const int nlo = lane & 15;
#pragma unroll
  for (int tm = 0; tm < 2; ++tm) {
#pragma unroll
    for (int tn = 0; tn < TN; ++tn) {
      int gn = bn0 + wn*(TN*16) + tn*16 + nlo;
      if (gn >= N) continue;
      float bv = (!(flags & 2) && bias) ? bias[gn] : 0.f;
#pragma unroll
      for (int r = 0; r < 8; ++r) {
        int gm = bm0 + wm*32 + tm*16 + hh + r;
        if (gm >= M) continue;
        long idx = (long)gm * ldc + gn;
        float v = acc[tm][tn][r];
        if (flags & 2) {
          atomicAdd(&C[idx], v);
        } else {
          v += bv;
          if (flags & 1) v = fmaxf(v, 0.f);
          if (resid) v += resid[idx];
          C[idx] = v;
        }
      }
    }
  }
}

// ---------------------------------------------------------------------------
// Fused attention per (batch, head): S=128, HD=64.
// scores = (Q*0.125) K^T  (WMMA), softmax in registers (16-lane shfl_xor
// matching the WMMA C layout), P -> LDS f16 (aliased over Q/K), ctx = P V.
// ---------------------------------------------------------------------------
__global__ __launch_bounds__(256) void attn_kernel(const float* __restrict__ qkv,
                                                   float* __restrict__ ctx)
{
  extern __shared__ _Float16 smem[];
  _Float16* sQ  = smem;                // 128*72
  _Float16* sK  = smem + 128 * 72;     // 128*72
  _Float16* sVt = smem + 2 * 128 * 72; // 64*136, V^T (e-major)
  _Float16* sP  = smem;                // 128*136, aliases Q/K after phase 2

  const int head = blockIdx.x;
  const int b    = blockIdx.y;
  const long base = (long)b * 128 * 1920 + head * 64;

  for (int i = threadIdx.x; i < 128 * 64; i += 256) {
    int s = i >> 6, d = i & 63;
    const float* p = qkv + base + (long)s * 1920 + d;
    sQ[s*72 + d]   = (_Float16)(p[0] * 0.125f);   // HD^-0.5
    sK[s*72 + d]   = (_Float16)(p[640]);
    sVt[d*136 + s] = (_Float16)(p[1280]);
  }
  __syncthreads();

  const int w    = threadIdx.x >> 5;   // wave owns score rows 16w..16w+15
  const int lane = threadIdx.x & 31;
  const int nlo  = lane & 15;
  const int hh   = (lane & 16) ? 8 : 0;
  const v8f vz = {0.f,0.f,0.f,0.f,0.f,0.f,0.f,0.f};

  v8f acc[8];
#pragma unroll
  for (int j = 0; j < 8; ++j) acc[j] = vz;
#pragma unroll
  for (int kk = 0; kk < 64; kk += 32) {
    v16h a = lds_frag(&sQ[(w*16) * 72 + kk], 72);
#pragma unroll
    for (int j = 0; j < 8; ++j) {
      v16h bf = lds_frag(&sK[(j*16) * 72 + kk], 72);
      acc[j] = WMMA_F16(a, bf, acc[j]);
    }
  }
  // softmax: row (w*16+hh+r) is spread across 16 lanes (same half) x 8 tiles
#pragma unroll
  for (int r = 0; r < 8; ++r) {
    float m = -3.0e38f;
#pragma unroll
    for (int j = 0; j < 8; ++j) m = fmaxf(m, acc[j][r]);
#pragma unroll
    for (int x = 1; x < 16; x <<= 1) m = fmaxf(m, __shfl_xor(m, x, 32));
    float ssum = 0.f;
#pragma unroll
    for (int j = 0; j < 8; ++j) { float e = __expf(acc[j][r] - m); acc[j][r] = e; ssum += e; }
#pragma unroll
    for (int x = 1; x < 16; x <<= 1) ssum += __shfl_xor(ssum, x, 32);
    float inv = 1.f / ssum;
#pragma unroll
    for (int j = 0; j < 8; ++j) acc[j][r] *= inv;
  }
  __syncthreads();   // all waves done reading sQ/sK before sP overwrites them
#pragma unroll
  for (int r = 0; r < 8; ++r) {
    int row = w*16 + hh + r;
#pragma unroll
    for (int j = 0; j < 8; ++j) sP[row*136 + j*16 + nlo] = (_Float16)acc[j][r];
  }
  __syncthreads();

  v8f cacc[4];
#pragma unroll
  for (int j = 0; j < 4; ++j) cacc[j] = vz;
#pragma unroll
  for (int kk = 0; kk < 128; kk += 32) {
    v16h a = lds_frag(&sP[(w*16) * 136 + kk], 136);
#pragma unroll
    for (int j = 0; j < 4; ++j) {
      v16h bf = lds_frag(&sVt[(j*16) * 136 + kk], 136);
      cacc[j] = WMMA_F16(a, bf, cacc[j]);
    }
  }
#pragma unroll
  for (int j = 0; j < 4; ++j) {
    int col = head*64 + j*16 + nlo;
#pragma unroll
    for (int r = 0; r < 8; ++r) {
      int row = w*16 + hh + r;
      ctx[((long)(b*128 + row)) * 640 + col] = cacc[j][r];
    }
  }
}

// ---------------------------------------------------------------------------
// Small helper kernels
// ---------------------------------------------------------------------------
__device__ __forceinline__ float block_reduce_sum(float v, float* red) {
  int t = threadIdx.x;
  red[t] = v; __syncthreads();
  for (int off = 128; off > 0; off >>= 1) {
    if (t < off) red[t] += red[t + off];
    __syncthreads();
  }
  float r = red[0]; __syncthreads();
  return r;
}

// x_ln = LN(attn + h); scattered directly into xcat[b, s*640 + e]
__global__ __launch_bounds__(256) void add_ln_xcat(
    const float* __restrict__ attn, const float* __restrict__ hbuf,
    const float* __restrict__ g, const float* __restrict__ bt,
    float* __restrict__ xcat)
{
  __shared__ float red[256];
  const int s = blockIdx.x, b = blockIdx.y;
  const long row = (long)b * 128 + s;
  const float* pa = attn + row * 640;
  const float* ph = hbuf + row * 640;
  float v[3]; int c = 0; float sum = 0.f;
  for (int i = threadIdx.x; i < 640; i += 256) { float t = pa[i] + ph[i]; v[c++] = t; sum += t; }
  sum = block_reduce_sum(sum, red);
  const float mean = sum * (1.f / 640.f);
  float var = 0.f;
  for (int k = 0; k < c; ++k) { float d = v[k] - mean; var += d * d; }
  var = block_reduce_sum(var, red) * (1.f / 640.f);
  const float inv = rsqrtf(var + 1e-5f);
  float* out = xcat + (long)b * 82080 + s * 640;
  c = 0;
  for (int i = threadIdx.x; i < 640; i += 256) { out[i] = (v[c++] - mean) * inv * g[i] + bt[i]; }
}

__global__ __launch_bounds__(256) void ln_rows(
    const float* __restrict__ x, const float* __restrict__ g,
    const float* __restrict__ bt, float* __restrict__ y, int n)
{
  __shared__ float red[256];
  const long row = blockIdx.x;
  const float* px = x + row * n;
  float v[8]; int c = 0; float sum = 0.f;
  for (int i = threadIdx.x; i < n; i += 256) { float t = px[i]; v[c++] = t; sum += t; }
  sum = block_reduce_sum(sum, red);
  const float mean = sum / (float)n;
  float var = 0.f;
  for (int k = 0; k < c; ++k) { float d = v[k] - mean; var += d * d; }
  var = block_reduce_sum(var, red) / (float)n;
  const float inv = rsqrtf(var + 1e-5f);
  float* py = y + row * n;
  c = 0;
  for (int i = threadIdx.x; i < n; i += 256) { py[i] = (v[c++] - mean) * inv * g[i] + bt[i]; }
}

__global__ void xcat_tail(const float* __restrict__ la, float* __restrict__ xcat) {
  int b = blockIdx.x, t = threadIdx.x;
  if (t < 160) xcat[(long)b * 82080 + 81920 + t] = (t < 129) ? la[(long)b * 129 + t] : 0.f;
}

__global__ void xcat2_tail(const float* __restrict__ la, float* __restrict__ x) {
  int b = blockIdx.x, t = threadIdx.x;
  if (t < 159) x[(long)b * 288 + 129 + t] = (t < 129) ? la[(long)b * 129 + t] : 0.f;
}

__global__ void fill_zero(float* __restrict__ p, long n) {
  long i = (long)blockIdx.x * 256 + threadIdx.x;
  if (i < n) p[i] = 0.f;
}

__global__ void bias_relu(float* __restrict__ x, const float* __restrict__ bt,
                          int n, long total) {
  long i = (long)blockIdx.x * 256 + threadIdx.x;
  if (i < total) { float v = x[i] + bt[i % n]; x[i] = fmaxf(v, 0.f); }
}

// Exact top-64 (lax.top_k tie-break: lower index wins) + masked softmax.
__global__ __launch_bounds__(256) void topk_softmax(const float* __restrict__ sc,
                                                    float* __restrict__ out)
{
  __shared__ float s[129];
  __shared__ float red[256];
  int b = blockIdx.x, t = threadIdx.x;
  if (t < 129) s[t] = sc[(long)b * 129 + t];
  __syncthreads();
  bool win = false; float mine = 0.f;
  if (t < 129) {
    mine = s[t];
    int cnt = 0;
    for (int j = 0; j < 129; ++j) {
      float o = s[j];
      cnt += (o > mine || (o == mine && j < t)) ? 1 : 0;
    }
    win = (cnt < 64);
  }
  red[t] = win ? mine : -3.0e38f; __syncthreads();
  for (int off = 128; off > 0; off >>= 1) {
    if (t < off) red[t] = fmaxf(red[t], red[t + off]);
    __syncthreads();
  }
  float mx = red[0]; __syncthreads();
  float ex = win ? __expf(mine - mx) : 0.f;
  red[t] = ex; __syncthreads();
  for (int off = 128; off > 0; off >>= 1) {
    if (t < off) red[t] += red[t + off];
    __syncthreads();
  }
  float sum = red[0];
  if (t < 129) out[(long)b * 129 + t] = ex / sum;
}

// ---------------------------------------------------------------------------
// Orchestration
// ---------------------------------------------------------------------------
extern "C" void kernel_launch(void* const* d_in, const int* in_sizes, int n_in,
                              void* d_out, int out_size, void* d_ws, size_t ws_size,
                              hipStream_t stream) {
  (void)in_sizes; (void)n_in; (void)out_size; (void)ws_size;
  const float* hist      = (const float*)d_in[0];
  const float* last_act  = (const float*)d_in[1];
  const float* embed_w   = (const float*)d_in[2];
  const float* embed_b   = (const float*)d_in[3];
  const float* in_proj_w = (const float*)d_in[4];
  const float* in_proj_b = (const float*)d_in[5];
  const float* out_proj_w= (const float*)d_in[6];
  const float* out_proj_b= (const float*)d_in[7];
  const float* lnorm_g   = (const float*)d_in[8];
  const float* lnorm_b   = (const float*)d_in[9];
  const float* fc1_w     = (const float*)d_in[10];
  const float* fc1_b     = (const float*)d_in[11];
  const float* fc2_w     = (const float*)d_in[12];
  const float* fc2_b     = (const float*)d_in[13];
  const float* ln1_g     = (const float*)d_in[14];
  const float* ln1_b     = (const float*)d_in[15];
  const float* fc3_w     = (const float*)d_in[16];
  const float* fc3_b     = (const float*)d_in[17];
  const float* fc4_w     = (const float*)d_in[18];
  const float* fc4_b     = (const float*)d_in[19];
  const float* fc5_w     = (const float*)d_in[20];
  const float* fc5_b     = (const float*)d_in[21];
  const float* fc6_w     = (const float*)d_in[22];
  const float* fc6_b     = (const float*)d_in[23];

  float* ws = (float*)d_ws;
  size_t o = 0;
  float* hbuf = ws + o;  o += (size_t)65536 * 640;    // embedded h (kept for residual)
  float* qkvb = ws + o;  o += (size_t)65536 * 1920;   // qkv
  float* ctxb = ws + o;  o += (size_t)65536 * 640;    // attention context
  float* attnb = qkvb;                                 // reuse qkv space after attention
  float* xcat = ws + o;  o += (size_t)512 * 82080;    // [LN(attn+h) | last_action | 0-pad]
  float* x1   = ws + o;  o += (size_t)512 * 2048;
  float* x2   = ws + o;  o += (size_t)512 * 2048;
  float* x2n  = ws + o;  o += (size_t)512 * 2048;
  float* x3   = ws + o;  o += (size_t)512 * 1024;
  float* x4   = ws + o;  o += (size_t)512 * 512;
  float* xc2  = ws + o;  o += (size_t)512 * 288;      // [fc5 out | last_action | pad]
  float* scr  = ws + o;  o += (size_t)512 * 129;      // final scores

  const dim3 blk(256);

  // 1) h = hist @ embed_w^T + embed_b          [65536,640]
  wmma_gemm<4, 128><<<dim3(5, 512, 1), blk, 0, stream>>>(
      hist, 640, embed_w, 640, embed_b, nullptr, hbuf, 640,
      65536, 640, 640, 0, 640, 0);
  // 2) qkv = h @ in_proj_w^T + in_proj_b       [65536,1920]
  wmma_gemm<4, 128><<<dim3(15, 512, 1), blk, 0, stream>>>(
      hbuf, 640, in_proj_w, 640, in_proj_b, nullptr, qkvb, 1920,
      65536, 1920, 640, 0, 640, 0);
  // 3) fused attention per (head, batch)        ctx [65536,640]
  attn_kernel<<<dim3(10, 512, 1), blk, 54272, stream>>>(qkvb, ctxb);
  // 4) attn_out = ctx @ out_proj_w^T + b        (overwrites qkv space)
  wmma_gemm<4, 128><<<dim3(5, 512, 1), blk, 0, stream>>>(
      ctxb, 640, out_proj_w, 640, out_proj_b, nullptr, attnb, 640,
      65536, 640, 640, 0, 640, 0);
  // 5) xcat body = LN(attn_out + h), tail = last_action + zero pad
  add_ln_xcat<<<dim3(128, 512, 1), blk, 0, stream>>>(attnb, hbuf, lnorm_g, lnorm_b, xcat);
  xcat_tail<<<dim3(512), dim3(192), 0, stream>>>(last_act, xcat);
  // 6) fc1: split-K x8 atomic accumulate, then bias+ReLU epilogue
  fill_zero<<<dim3(4096), blk, 0, stream>>>(x1, (long)512 * 2048);
  wmma_gemm<1, 64><<<dim3(16, 8, 8), blk, 0, stream>>>(
      xcat, 82080, fc1_w, 82049, nullptr, nullptr, x1, 2048,
      512, 2048, 82049, 0, 10272, 2);
  bias_relu<<<dim3(4096), blk, 0, stream>>>(x1, fc1_b, 2048, (long)512 * 2048);
  // 7) x2 = relu(x1 @ fc2_w^T + b) + x1
  wmma_gemm<4, 64><<<dim3(16, 8, 1), blk, 0, stream>>>(
      x1, 2048, fc2_w, 2048, fc2_b, x1, x2, 2048,
      512, 2048, 2048, 0, 2048, 1);
  // 8) ln1
  ln_rows<<<dim3(512), blk, 0, stream>>>(x2, ln1_g, ln1_b, x2n, 2048);
  // 9) fc3 / fc4 (ReLU)
  wmma_gemm<4, 64><<<dim3(8, 8, 1), blk, 0, stream>>>(
      x2n, 2048, fc3_w, 2048, fc3_b, nullptr, x3, 1024,
      512, 1024, 2048, 0, 2048, 1);
  wmma_gemm<4, 64><<<dim3(4, 8, 1), blk, 0, stream>>>(
      x3, 1024, fc4_w, 1024, fc4_b, nullptr, x4, 512,
      512, 512, 1024, 0, 1024, 1);
  // 10) fc5 -> cols [0,129) of xc2, then append last_action + pad
  wmma_gemm<4, 64><<<dim3(2, 8, 1), blk, 0, stream>>>(
      x4, 512, fc5_w, 512, fc5_b, nullptr, xc2, 288,
      512, 129, 512, 0, 512, 0);
  xcat2_tail<<<dim3(512), dim3(192), 0, stream>>>(last_act, xc2);
  // 11) fc6 -> final_scores [512,129]
  wmma_gemm<1, 64><<<dim3(2, 8, 1), blk, 0, stream>>>(
      xc2, 288, fc6_w, 258, fc6_b, nullptr, scr, 129,
      512, 129, 258, 0, 288, 0);
  // 12) exact top-64 winners mask + masked softmax
  topk_softmax<<<dim3(512), blk, 0, stream>>>(scr, (float*)d_out);
}
